// PocketGraphEncoder_36086315221251
// MI455X (gfx1250) — compile-verified, matrix operand order checked by reading, not run on hardware
//
#include <hip/hip_runtime.h>
#include <hip/hip_bf16.h>

typedef __attribute__((ext_vector_type(16))) __bf16 v16bf;
typedef __attribute__((ext_vector_type(8)))  __bf16 v8bf;
typedef __attribute__((ext_vector_type(8)))  float  v8f;

#define IN_DIM 21
#define HID    256
#define KTOT   512   // concatenated [x | agg] feature dim

// ---------------- K1: x = relu(node_scalar @ W_in + b_in) -> xa[:, 0:256] (bf16)
__global__ void k_input_linear(const float* __restrict__ ns,
                               const float* __restrict__ W_in,
                               const float* __restrict__ b_in,
                               __bf16* __restrict__ xa) {
  __shared__ float row[IN_DIM];
  const int node = blockIdx.x;
  const int t = threadIdx.x;
  if (t < IN_DIM) row[t] = ns[(size_t)node * IN_DIM + t];
  __syncthreads();
  float acc = b_in[t];
#pragma unroll
  for (int k = 0; k < IN_DIM; ++k)
    acc = fmaf(row[k], W_in[k * HID + t], acc);
  acc = fmaxf(acc, 0.0f);
  xa[(size_t)node * KTOT + t] = (__bf16)acc;
}

// ---------------- K2: pack [W_self; W_nei] (512x256) into WMMA bf16 B-fragment layout.
// Dest layout: Bpack[((ntile*16 + kstep)*32 + lane)*16 + j]
// lane = 16*half + n_local ; j<8 -> K = kstep*32 + half*8 + j ; j>=8 -> +16
__global__ void k_pack_B(const float* __restrict__ Wself,
                         const float* __restrict__ Wnei,
                         __bf16* __restrict__ Bpack) {
  const int idx   = blockIdx.x * 256 + threadIdx.x;   // 0 .. 131071
  const int j     = idx & 15;
  const int lane  = (idx >> 4) & 31;
  const int kstep = (idx >> 9) & 15;
  const int ntile = idx >> 13;
  const int half  = lane >> 4;
  const int n     = ntile * 16 + (lane & 15);
  const int K     = kstep * 32 + ((j < 8) ? (half * 8 + j) : (16 + half * 8 + (j - 8)));
  const float v   = (K < HID) ? Wself[(size_t)K * HID + n]
                              : Wnei[(size_t)(K - HID) * HID + n];
  Bpack[idx] = (__bf16)v;
}

// ---------------- K3: per-edge scatter-add (one wave32 per edge), f32 accumulation
__global__ void k_edge_scatter(const long long* __restrict__ ei,
                               const __bf16* __restrict__ xa,
                               float* __restrict__ agg,
                               float* __restrict__ deg,
                               long long E) {
  const int wave = threadIdx.x >> 5;
  const int lane = threadIdx.x & 31;
  const long long e = (long long)blockIdx.x * 8 + wave;
  if (e >= E) return;
  const long long s = ei[e];
  const long long d = ei[E + e];
  const __bf16* __restrict__ xs = xa + (size_t)s * KTOT;
  float* __restrict__ ad = agg + (size_t)d * HID;
#pragma unroll
  for (int i = 0; i < 8; ++i) {
    const int c = lane + i * 32;
    atomicAdd(&ad[c], (float)xs[c]);
  }
  if (lane == 0) atomicAdd(&deg[d], 1.0f);
}

// ---------------- K4: agg / max(deg,1) -> xa[:, 256:512] (bf16)
__global__ void k_normalize(const float* __restrict__ agg,
                            const float* __restrict__ deg,
                            __bf16* __restrict__ xa) {
  const int node = blockIdx.x;
  const int c = threadIdx.x;
  const float inv = 1.0f / fmaxf(deg[node], 1.0f);
  xa[(size_t)node * KTOT + HID + c] = (__bf16)(agg[(size_t)node * HID + c] * inv);
}

// A-fragment loader: two contiguous 16B chunks per lane per K-step (ISA 16-bit A layout)
static __device__ __forceinline__ v16bf load_a_frag(const __bf16* __restrict__ Arow,
                                                    int ks, int half) {
  const __bf16* ap = Arow + ks * 32 + half * 8;
  v8bf lo = *(const v8bf*)(ap);
  v8bf hi = *(const v8bf*)(ap + 16);
  v16bf a;
#pragma unroll
  for (int i = 0; i < 8; ++i) { a[i] = lo[i]; a[8 + i] = hi[i]; }
  return a;
}

// ---------------- K5: fused GEMM h = relu([x|agg] @ [Wself;Wnei] + bias) + mean pool.
// One wave per 16-row M-strip: A fragments loaded ONCE per K-step and reused across
// all 16 column tiles (16 v8f accumulators resident = 128 VGPRs). A and B fragments
// are software double-buffered, and sched_group_barrier pins the interleave
// [4 vmem] (1 wmma, 2 vmem)x15 (1 wmma) so B(nt+1) loads fly during WMMA(nt)
// instead of a full s_wait_loadcnt 0 before every matrix op.
__global__ void __launch_bounds__(256)
k_gemm_wmma(const __bf16* __restrict__ xa,
            const __bf16* __restrict__ Bpack,
            const float* __restrict__ b_self,
            const float* __restrict__ b_nei,
            float* __restrict__ g_sum, int N) {
  const int wave  = threadIdx.x >> 5;
  const int lane  = threadIdx.x & 31;
  const int mtile = blockIdx.x * 8 + wave;     // 16-row strip id (wave-uniform)
  if (mtile * 16 >= N) return;                 // uniform: EXEC stays all-1s

  const int half = lane >> 4;
  const int lm   = lane & 15;

  // A: lane holds row (mtile*16+lm); per K-step chunks at K = ks*32 + half*8 (+0/+16)
  const __bf16* __restrict__ Arow = xa + (size_t)(mtile * 16 + lm) * KTOT;
  // B: packed fragment base for this lane; element offset of (ntile, kstep) is
  //    ((ntile*16 + kstep)*32 + lane)*16  = lane*16 + kstep*512 + ntile*8192
  const __bf16* __restrict__ Bl = Bpack + (size_t)lane * 16;

  v8f acc[16];
#pragma unroll
  for (int nt = 0; nt < 16; ++nt) acc[nt] = (v8f){};

  v16bf acur = load_a_frag(Arow, 0, half);
  for (int ks = 0; ks < 16; ++ks) {
    // prefetch next K-step's A fragment while this step's WMMAs run
    v16bf anext = acur;
    if (ks < 15) anext = load_a_frag(Arow, ks + 1, half);

    const __bf16* bp = Bl + (size_t)ks * 512;
    v16bf bcur = *(const v16bf*)(bp);
#pragma unroll
    for (int nt = 0; nt < 16; ++nt) {
      // prefetch next column tile's B fragment before issuing this WMMA
      v16bf bnext = bcur;
      if (nt < 15) bnext = *(const v16bf*)(bp + (size_t)(nt + 1) * 8192);
      acc[nt] = __builtin_amdgcn_wmma_f32_16x16x32_bf16(
          /*neg_a=*/false, acur, /*neg_b=*/false, bcur,
          /*c_mod=*/(short)0, acc[nt], /*reuse_a=*/false, /*reuse_b=*/false);
      bcur = bnext;
    }
    acur = anext;

#if defined(__has_builtin)
#if __has_builtin(__builtin_amdgcn_sched_group_barrier)
    // Pin the software pipeline: 0x20 = VMEM read, 0x08 = MFMA/WMMA.
    // Body holds 34 vmem reads (A-next pair + 16 B fragments x2 b128) + 16 WMMA.
    __builtin_amdgcn_sched_group_barrier(0x020, 4, 0);   // b0 pair + A-next pair
#pragma unroll
    for (int i = 0; i < 15; ++i) {
      __builtin_amdgcn_sched_group_barrier(0x008, 1, 0); // WMMA(nt)
      __builtin_amdgcn_sched_group_barrier(0x020, 2, 0); // B(nt+1) pair in flight
    }
    __builtin_amdgcn_sched_group_barrier(0x008, 1, 0);   // final WMMA
#endif
#endif
  }

  // C/D layout: lane (half, lm) holds column ncol; VGPR r holds row r + 8*half.
#pragma unroll
  for (int nt = 0; nt < 16; ++nt) {
    const int ncol = nt * 16 + lm;
    const float bias = b_self[ncol] + b_nei[ncol];
    float s = 0.0f;
#pragma unroll
    for (int r = 0; r < 8; ++r) s += fmaxf(acc[nt][r] + bias, 0.0f);
    // lanes l and l+16 hold the same column (complementary row halves): fold halves
    s += __shfl_xor(s, 16, 32);
    if (half == 0) atomicAdd(&g_sum[ncol], s);  // fused mean-pool partial (16 rows)
  }
}

// ---------------- K6: out = (g_sum / N) @ W_out + b_out   (256x256 GEMV, one block)
__global__ void k_out(const float* __restrict__ g_sum,
                      const float* __restrict__ W_out,
                      const float* __restrict__ b_out,
                      float* __restrict__ out, float invN) {
  const int t = threadIdx.x;
  float acc = b_out[t];
  for (int k = 0; k < HID; ++k)
    acc = fmaf(g_sum[k] * invN, W_out[(size_t)k * HID + t], acc);
  out[t] = acc;
}

extern "C" void kernel_launch(void* const* d_in, const int* in_sizes, int n_in,
                              void* d_out, int out_size, void* d_ws, size_t ws_size,
                              hipStream_t stream) {
  const float*     ns     = (const float*)d_in[0];
  const float*     W_in   = (const float*)d_in[1];
  const float*     b_in   = (const float*)d_in[2];
  const float*     W_self = (const float*)d_in[3];
  const float*     b_self = (const float*)d_in[4];
  const float*     W_nei  = (const float*)d_in[5];
  const float*     b_nei  = (const float*)d_in[6];
  const float*     W_out  = (const float*)d_in[7];
  const float*     b_out  = (const float*)d_in[8];
  const long long* ei     = (const long long*)d_in[9];
  float* out = (float*)d_out;

  const int       N = in_sizes[0] / IN_DIM;
  const long long E = (long long)in_sizes[9] / 2;

  // ---- workspace carve-up (256B aligned)
  char* ws = (char*)d_ws;
  size_t off = 0;
  __bf16* xa    = (__bf16*)(ws + off); off += (size_t)N * KTOT * 2; off = (off + 255) & ~(size_t)255;
  float*  agg   = (float*)(ws + off);  off += (size_t)N * HID * 4;  off = (off + 255) & ~(size_t)255;
  float*  deg   = (float*)(ws + off);  off += (size_t)N * 4;        off = (off + 255) & ~(size_t)255;
  __bf16* Bpack = (__bf16*)(ws + off); off += (size_t)16 * 16 * 32 * 16 * 2; off = (off + 255) & ~(size_t)255;
  float*  g_sum = (float*)(ws + off);  off += (size_t)HID * 4;

  // ---- zero accumulators (graph-capture-safe memset nodes)
  hipMemsetAsync(agg,   0, (size_t)N * HID * 4, stream);
  hipMemsetAsync(deg,   0, (size_t)N * 4,       stream);
  hipMemsetAsync(g_sum, 0, (size_t)HID * 4,     stream);

  // K1: input linear + relu -> bf16
  k_input_linear<<<N, HID, 0, stream>>>(ns, W_in, b_in, xa);
  // K2: pack weights into WMMA B-fragment layout (512 blocks * 256 = 131072 elems)
  k_pack_B<<<512, 256, 0, stream>>>(W_self, W_nei, Bpack);
  // K3: edge gather + scatter-add (one wave per edge)
  k_edge_scatter<<<(unsigned)((E + 7) / 8), 256, 0, stream>>>(ei, xa, agg, deg, E);
  // K4: degree-normalize -> bf16 second half
  k_normalize<<<N, HID, 0, stream>>>(agg, deg, xa);
  // K5: fused WMMA GEMM + bias + relu + mean-pool partials (one wave per M-strip)
  {
    const int mtiles = (N + 15) / 16;
    const int blocks = (mtiles + 7) / 8;
    k_gemm_wmma<<<blocks, 256, 0, stream>>>(xa, Bpack, b_self, b_nei, g_sum, N);
  }
  // K6: output head
  k_out<<<1, HID, 0, stream>>>(g_sum, W_out, b_out, out, 1.0f / (float)N);
}
